// AtomMPNN_43954695308037
// MI455X (gfx1250) — matrix operand order repaired and checked by reading, not compiled, not run on hardware
//
#include <hip/hip_runtime.h>

// =====================================================================
// AtomMPNN on MI455X (gfx1250), wave32 + WMMA bf16.
//
// Roofline: 68.9 GFLOP edge-MLP vs 134 MB bf16 gather @ 23.3 TB/s
// (~5.8us) -> compute/memory balanced at bf16 WMMA rates; all three
// GEMMs use v_wmma_f32_16x16x32_bf16 with f32 accumulators.
// GELU uses a branch-free A&S-7.1.26 erf (|err|<1.5e-7, below bf16
// noise) built on v_rcp_f32/v_exp_f32 so the VALU tail doesn't
// dominate the matrix pipe (libm erff emits exec-divergent branches).
// =====================================================================

#define Bc 4
#define Nc 4096
#define Kc 32
#define Hc 128

typedef __attribute__((ext_vector_type(16))) __bf16      v16bf;
typedef __attribute__((ext_vector_type(8)))  float       v8f;
typedef __attribute__((ext_vector_type(4)))  unsigned int u32x4;

union Frag { v16bf bf; u32x4 q[2]; };

static __device__ __forceinline__ unsigned short f2bf(float x) {
  unsigned u = __float_as_uint(x);
  u += 0x7FFFu + ((u >> 16) & 1u);          // round-to-nearest-even
  return (unsigned short)(u >> 16);
}

// Branch-free erf, Abramowitz&Stegun 7.1.26 (max abs err 1.5e-7).
static __device__ __forceinline__ float erf_fast(float x) {
  float a = __builtin_fabsf(x);
  float t = __builtin_amdgcn_rcpf(__builtin_fmaf(0.3275911f, a, 1.0f));
  float p = __builtin_fmaf(t, 1.061405429f, -1.453152027f);
  p = __builtin_fmaf(t, p, 1.421413741f);
  p = __builtin_fmaf(t, p, -0.284496736f);
  p = __builtin_fmaf(t, p, 0.254829592f);
  p = p * t;
  float e = __builtin_amdgcn_exp2f(a * a * -1.4426950408889634f); // exp(-a^2)
  float r = 1.0f - p * e;
  return __builtin_copysignf(r, x);
}

static __device__ __forceinline__ float gelu(float x) {
  return 0.5f * x * (1.0f + erf_fast(x * 0.70710678118654752f));
}

// ---------------- workspace layout (bytes) ----------------
// enc_bf16 : B*N*H   ushort  = 4,194,304
// W0t      : 128*264 ushort  =    67,584   (257-row W0 transposed, padded to 264)
// W1t      : 128*128 ushort  =    32,768
// W2t      : 128*128 ushort  =    32,768
// msg/upd  : B*N*H   float   = 8,388,608
// sum      : B*H     float   =     2,048
// sumsq    : B*H     float   =     2,048
// cnt      : B       float   =        16
#define OFF_ENCB  ((size_t)0)
#define OFF_W0T   ((size_t)4194304)
#define OFF_W1T   ((size_t)4261888)
#define OFF_W2T   ((size_t)4294656)
#define OFF_MSG   ((size_t)4327424)
#define OFF_SUM   ((size_t)12716032)
#define OFF_SUMSQ ((size_t)12718080)
#define OFF_CNT   ((size_t)12720128)
#define WS_NEEDED ((size_t)12720144)

// ---------------- kernels ----------------

__global__ void zero_kernel(float* p, int n) {
  int i = blockIdx.x * blockDim.x + threadIdx.x;
  if (i < n) p[i] = 0.f;
}

// Pre-mask node features -> bf16; transpose weights -> bf16 column-major
// (so each lane's B-fragment is two contiguous 16B chunks).
__global__ void prep_kernel(const float* __restrict__ enc,
                            const float* __restrict__ mask,
                            const float* __restrict__ W0,
                            const float* __restrict__ W1,
                            const float* __restrict__ W2,
                            unsigned short* __restrict__ encb,
                            unsigned short* __restrict__ w0t,
                            unsigned short* __restrict__ w1t,
                            unsigned short* __restrict__ w2t) {
  int idx = blockIdx.x * blockDim.x + threadIdx.x;
  const int RENC = Bc * Nc * Hc;     // 2,097,152
  const int R0   = 128 * 257;        //    32,896
  const int R1   = 128 * 128;        //    16,384
  if (idx < RENC) {
    encb[idx] = f2bf(enc[idx] * mask[idx >> 7]);
  } else if ((idx -= RENC) < R0) {
    int h = idx / 257, i = idx % 257;
    w0t[h * 264 + i] = f2bf(W0[i * Hc + h]);
  } else if ((idx -= R0) < R1) {
    int h = idx >> 7, i = idx & 127;
    w1t[h * 128 + i] = f2bf(W1[i * Hc + h]);
  } else if ((idx -= R1) < R1) {
    int h = idx >> 7, i = idx & 127;
    w2t[h * 128 + i] = f2bf(W2[i * Hc + h]);
  }
}

// Edge MLP: block = 128 threads (4 wave32) = 2 nodes = 64 edge rows.
// Each wave owns a 16-row M-tile and runs 16x16x32 bf16 WMMAs.
__global__ __launch_bounds__(128) void mlp_kernel(
    const unsigned short* __restrict__ encb,
    const unsigned short* __restrict__ w0t,
    const unsigned short* __restrict__ w1t,
    const unsigned short* __restrict__ w2t,
    const float* __restrict__ W0f,
    const float* __restrict__ b0,
    const float* __restrict__ b1,
    const float* __restrict__ b2,
    const int*   __restrict__ eidx,
    const float* __restrict__ dist,
    float* __restrict__ msg) {
  __shared__ alignas(16) unsigned short A0[64 * 256];   // [row][src(128)|self(128)]
  __shared__ alignas(16) unsigned short Hb[64 * 128];   // hidden activations
  __shared__ int   sIdx[64];
  __shared__ float sDist[64];
  __shared__ float sValid[64];
  __shared__ float sNodeSum[2 * Hc];
  __shared__ int   sVcnt[2];

  const int tid = threadIdx.x;

  sNodeSum[tid] = 0.f;
  sNodeSum[tid + 128] = 0.f;
  if (tid < 2) sVcnt[tid] = 0;
  __syncthreads();

  // stage edge metadata
  if (tid < 64) {
    int nl = tid >> 5, k = tid & 31;
    int g = blockIdx.x * 2 + nl;                  // flat node id (b*N + n)
    int e = eidx[g * Kc + k];
    int valid = (e != -1);
    sIdx[tid]   = valid ? e : 0;
    sValid[tid] = valid ? 1.f : 0.f;
    sDist[tid]  = dist[g * Kc + k];
    if (valid) atomicAdd(&sVcnt[nl], 1);
  }
  __syncthreads();

  // gather bf16 neighbor+self rows into LDS (16B chunks)
  for (int t = tid; t < 64 * 32; t += 128) {
    int row = t >> 5, q = t & 31;
    int nl = row >> 5;
    int g = blockIdx.x * 2 + nl;
    int bbase = g & ~(Nc - 1);                    // b*N
    const u32x4* src;
    unsigned dsto;
    if (q < 16) {
      src  = (const u32x4*)&encb[(size_t)(bbase + sIdx[row]) * Hc + q * 8];
      dsto = row * 256 + q * 8;
    } else {
      int qq = q - 16;
      src  = (const u32x4*)&encb[(size_t)g * Hc + qq * 8];
      dsto = row * 256 + 128 + qq * 8;
    }
    *(u32x4*)&A0[dsto] = *src;
  }
  __syncthreads();

  const int wave = tid >> 5, lane = tid & 31;
  const int hi = lane >> 4, lo = lane & 15;
  const int rowA = wave * 16 + lo;

  float dist8[8], vm8[8];
#pragma unroll
  for (int r = 0; r < 8; ++r) {
    int m = wave * 16 + r + 8 * hi;               // C-layout row for this lane/VGPR
    dist8[r] = sDist[m];
    vm8[r]   = sValid[m];
  }

  // ---------- layer 0 : 16x256 @ 256x128, + rank-1 dist term ----------
  Frag a0[8];
#pragma unroll
  for (int ks = 0; ks < 8; ++ks) {
    int ko = ks * 32;
    a0[ks].q[0] = *(const u32x4*)&A0[rowA * 256 + ko + 8 * hi];
    a0[ks].q[1] = *(const u32x4*)&A0[rowA * 256 + ko + 16 + 8 * hi];
  }
#pragma unroll
  for (int j = 0; j < 8; ++j) {
    v8f c = {};
#pragma unroll
    for (int ks = 0; ks < 8; ++ks) {
      Frag w;
      const unsigned short* wp = &w0t[(size_t)(j * 16 + lo) * 264 + ks * 32 + 8 * hi];
      w.q[0] = *(const u32x4*)(wp);
      w.q[1] = *(const u32x4*)(wp + 16);
      c = __builtin_amdgcn_wmma_f32_16x16x32_bf16(false, a0[ks].bf, false, w.bf,
                                                  (short)0, c, false, false);
    }
    int n_ = j * 16 + lo;
    float wl = W0f[256 * Hc + n_];                // 257th input row (distance)
    float bb = b0[n_];
#pragma unroll
    for (int r = 0; r < 8; ++r) {
      float x = gelu(c[r] + dist8[r] * wl + bb);
      Hb[(wave * 16 + r + 8 * hi) * Hc + n_] = f2bf(x);
    }
  }
  __syncthreads();

  // ---------- layer 1 : 16x128 @ 128x128 ----------
  Frag a1[4];
#pragma unroll
  for (int ks = 0; ks < 4; ++ks) {
    int ko = ks * 32;
    a1[ks].q[0] = *(const u32x4*)&Hb[rowA * 128 + ko + 8 * hi];
    a1[ks].q[1] = *(const u32x4*)&Hb[rowA * 128 + ko + 16 + 8 * hi];
  }
#pragma unroll
  for (int j = 0; j < 8; ++j) {
    v8f c = {};
#pragma unroll
    for (int ks = 0; ks < 4; ++ks) {
      Frag w;
      const unsigned short* wp = &w1t[(size_t)(j * 16 + lo) * 128 + ks * 32 + 8 * hi];
      w.q[0] = *(const u32x4*)(wp);
      w.q[1] = *(const u32x4*)(wp + 16);
      c = __builtin_amdgcn_wmma_f32_16x16x32_bf16(false, a1[ks].bf, false, w.bf,
                                                  (short)0, c, false, false);
    }
    int n_ = j * 16 + lo;
    float bb = b1[n_];
#pragma unroll
    for (int r = 0; r < 8; ++r) {
      float x = gelu(c[r] + bb);
      Hb[(wave * 16 + r + 8 * hi) * Hc + n_] = f2bf(x);
    }
  }
  __syncthreads();

  // ---------- layer 2 : 16x128 @ 128x128, mask + aggregate ----------
  Frag a2[4];
#pragma unroll
  for (int ks = 0; ks < 4; ++ks) {
    int ko = ks * 32;
    a2[ks].q[0] = *(const u32x4*)&Hb[rowA * 128 + ko + 8 * hi];
    a2[ks].q[1] = *(const u32x4*)&Hb[rowA * 128 + ko + 16 + 8 * hi];
  }
#pragma unroll
  for (int j = 0; j < 8; ++j) {
    v8f c = {};
#pragma unroll
    for (int ks = 0; ks < 4; ++ks) {
      Frag w;
      const unsigned short* wp = &w2t[(size_t)(j * 16 + lo) * 128 + ks * 32 + 8 * hi];
      w.q[0] = *(const u32x4*)(wp);
      w.q[1] = *(const u32x4*)(wp + 16);
      c = __builtin_amdgcn_wmma_f32_16x16x32_bf16(false, a2[ks].bf, false, w.bf,
                                                  (short)0, c, false, false);
    }
    int n_ = j * 16 + lo;
    float bb = b2[n_];
    float p = 0.f;
#pragma unroll
    for (int r = 0; r < 8; ++r) {
      p += gelu(c[r] + bb) * vm8[r];              // valid-masked message
    }
    atomicAdd(&sNodeSum[(wave >> 1) * Hc + n_], p);   // ds_add_f32
  }
  __syncthreads();

  // masked mean over neighbors -> msg buffer
#pragma unroll
  for (int nl = 0; nl < 2; ++nl) {
    int g = blockIdx.x * 2 + nl;
    float cnt = (float)sVcnt[nl];
    if (cnt == 0.f) cnt = 1.f;
    msg[(size_t)g * Hc + tid] = sNodeSum[nl * Hc + tid] / cnt;
  }
}

// upd = (atom_encode + msg)*m  (in place over msg buffer), accumulate per-(b,h)
// sums/sumsq and per-b valid-node counts for graph norm.
__global__ __launch_bounds__(128) void stats_kernel(
    const float* __restrict__ enc, const float* __restrict__ mask,
    float* __restrict__ upd, float* __restrict__ gsum,
    float* __restrict__ gsumsq, float* __restrict__ gcnt) {
  int blk = blockIdx.x;
  int b = blk >> 5, chunk = blk & 31;             // 32 chunks of 128 nodes
  int h = threadIdx.x;
  float s = 0.f, s2 = 0.f, cm = 0.f;
  for (int i = 0; i < 128; ++i) {
    int n = chunk * 128 + i;
    size_t base = ((size_t)(b * Nc + n)) * Hc + h;
    float mval = mask[b * Nc + n];
    float u = (enc[base] + upd[base]) * mval;
    upd[base] = u;
    s += u; s2 += u * u;
    if (h == 0) cm += mval;
  }
  atomicAdd(&gsum[b * Hc + h], s);
  atomicAdd(&gsumsq[b * Hc + h], s2);
  if (h == 0) atomicAdd(&gcnt[b], cm);
}

// Graph norm: var matches reference exactly: sum over ALL N of (mf-mean)^2 / cnt.
__global__ __launch_bounds__(128) void final_kernel(
    const float* __restrict__ mask, const float* __restrict__ upd,
    const float* __restrict__ gsum, const float* __restrict__ gsumsq,
    const float* __restrict__ gcnt, const float* __restrict__ scale,
    const float* __restrict__ shift, float* __restrict__ out) {
  int g = blockIdx.x;
  int h = threadIdx.x;
  int b = g >> 12;
  float cnt = gcnt[b];
  if (cnt == 0.f) cnt = 1.f;
  float s = gsum[b * Hc + h], s2 = gsumsq[b * Hc + h];
  float mean = s / cnt;
  float var = (s2 - 2.f * mean * s + (float)Nc * mean * mean) / cnt;
  float stdv = sqrtf(var + 1e-5f);
  float u = upd[(size_t)g * Hc + h];
  out[(size_t)g * Hc + h] = ((u - mean) / stdv * scale[h] + shift[h]) * mask[g];
}

// ---------------- launch ----------------
extern "C" void kernel_launch(void* const* d_in, const int* in_sizes, int n_in,
                              void* d_out, int out_size, void* d_ws, size_t ws_size,
                              hipStream_t stream) {
  (void)in_sizes; (void)n_in; (void)out_size;
  if (ws_size < WS_NEEDED) return;                // need ~12.2 MB scratch

  const float* atom_encode = (const float*)d_in[0];
  const float* atom_mask   = (const float*)d_in[1];
  const float* dist        = (const float*)d_in[2];
  const int*   eidx        = (const int*)d_in[3];
  const float* W0 = (const float*)d_in[4];
  const float* b0 = (const float*)d_in[5];
  const float* W1 = (const float*)d_in[6];
  const float* b1 = (const float*)d_in[7];
  const float* W2 = (const float*)d_in[8];
  const float* b2 = (const float*)d_in[9];
  const float* scale = (const float*)d_in[10];
  const float* shift = (const float*)d_in[11];

  char* ws = (char*)d_ws;
  unsigned short* encb = (unsigned short*)(ws + OFF_ENCB);
  unsigned short* w0t  = (unsigned short*)(ws + OFF_W0T);
  unsigned short* w1t  = (unsigned short*)(ws + OFF_W1T);
  unsigned short* w2t  = (unsigned short*)(ws + OFF_W2T);
  float* msg    = (float*)(ws + OFF_MSG);
  float* gsum   = (float*)(ws + OFF_SUM);
  float* gsumsq = (float*)(ws + OFF_SUMSQ);
  float* gcnt   = (float*)(ws + OFF_CNT);
  float* out    = (float*)d_out;

  // 1) zero reduction buffers (sum+sumsq+cnt = 1028 floats, contiguous)
  zero_kernel<<<5, 256, 0, stream>>>(gsum, 1028);

  // 2) bf16 pre-mask + weight transpose
  const int PREP_TOTAL = Bc * Nc * Hc + 128 * 257 + 2 * 128 * 128;
  prep_kernel<<<(PREP_TOTAL + 255) / 256, 256, 0, stream>>>(
      atom_encode, atom_mask, W0, W1, W2, encb, w0t, w1t, w2t);

  // 3) WMMA edge MLP + masked mean aggregation (2 nodes / block)
  mlp_kernel<<<Bc * Nc / 2, 128, 0, stream>>>(
      encb, w0t, w1t, w2t, W0, b0, b1, b2, eidx, dist, msg);

  // 4) residual update + per-batch stats
  stats_kernel<<<Bc * 32, 128, 0, stream>>>(atom_encode, atom_mask, msg,
                                            gsum, gsumsq, gcnt);

  // 5) graph norm finalize
  final_kernel<<<Bc * Nc, 128, 0, stream>>>(atom_mask, msg, gsum, gsumsq,
                                            gcnt, scale, shift, out);

  // 6) tuple tail: (atom_mask, dist_neighbors, edge_index) copied flat
  size_t o = (size_t)Bc * Nc * Hc;
  hipMemcpyAsync(out + o, d_in[1], (size_t)Bc * Nc * 4,
                 hipMemcpyDeviceToDevice, stream);
  o += (size_t)Bc * Nc;
  hipMemcpyAsync(out + o, d_in[2], (size_t)Bc * Nc * Kc * 4,
                 hipMemcpyDeviceToDevice, stream);
  o += (size_t)Bc * Nc * Kc;
  hipMemcpyAsync(out + o, d_in[3], (size_t)Bc * Nc * Kc * 4,
                 hipMemcpyDeviceToDevice, stream);
}
